// ImitativeModel_47734266528051
// MI455X (gfx1250) — matrix operand order repaired and checked by reading, not compiled, not run on hardware
//
#include <hip/hip_runtime.h>
#include <hip/hip_bf16.h>

typedef __bf16 bf16_t;
typedef __attribute__((ext_vector_type(16))) bf16_t v16bf;
typedef __attribute__((ext_vector_type(8)))  float  v8f;

#define BATCH 131072
#define TT 16
#define DD 2
#define HH 64
#define GG 192            // 3*H

#define WAVES 8
#define RPW 16            // batch rows per wave
#define HS_STRIDE 68      // f32 staging stride for h  (16 x 64 tile)
#define HID_STRIDE 36     // f32 staging stride for hid (16 x 32 tile)
#define WH_STRIDE 72      // bf16 row stride for W_hh in LDS (192 x 64)
#define W1_STRIDE 72      // bf16 row stride for W1^T in LDS (32 x 64)

// Load a 32x16 bf16 B-fragment (8 VGPRs) from LDS. Caller passes the lane's
// row pointer (row = N = lane%16 within the tile) at the lane's K base
// (kc*32 + 8*(lane/16)); elements 0-7 hold K=base+0..7, 8-15 hold K=base+16..23.
__device__ __forceinline__ v16bf ld_bfrag(const bf16_t* p) {
  union { uint4 u[2]; v16bf v; } c;
  c.u[0] = *(const uint4*)(p);
  c.u[1] = *(const uint4*)(p + 16);
  return c.v;
}

// Build a 16x32 bf16 A-fragment (8 VGPRs) from the f32 LDS staging row of
// this lane's M row. Per ISA 7.12.2 (16-bit A): lane half hf contributes
// K = kc*32 + 8*hf + {0..7} in VGPRs 0-3 and +16 in VGPRs 4-7.
__device__ __forceinline__ v16bf mkA(const float* hrow, int kc, int hf) {
  const int base = kc*32 + 8*hf;
  const float4 q0 = *(const float4*)(hrow + base);
  const float4 q1 = *(const float4*)(hrow + base + 4);
  const float4 q2 = *(const float4*)(hrow + base + 16);
  const float4 q3 = *(const float4*)(hrow + base + 20);
  v16bf a;
  a[0] = (bf16_t)q0.x;  a[1] = (bf16_t)q0.y;  a[2] = (bf16_t)q0.z;  a[3] = (bf16_t)q0.w;
  a[4] = (bf16_t)q1.x;  a[5] = (bf16_t)q1.y;  a[6] = (bf16_t)q1.z;  a[7] = (bf16_t)q1.w;
  a[8] = (bf16_t)q2.x;  a[9] = (bf16_t)q2.y;  a[10] = (bf16_t)q2.z; a[11] = (bf16_t)q2.w;
  a[12] = (bf16_t)q3.x; a[13] = (bf16_t)q3.y; a[14] = (bf16_t)q3.z; a[15] = (bf16_t)q3.w;
  return a;
}

__device__ __forceinline__ v8f wmma_bf16(v16bf a, v16bf b, v8f c) {
  return __builtin_amdgcn_wmma_f32_16x16x32_bf16(false, a, false, b, (short)0, c,
                                                 false, false);
}

// Fast transcendentals: v_exp_f32 / v_rcp_f32 / v_log_f32 only (TRANS ops
// co-execute with the WMMA pipeline; avoids the IEEE div-fixup sequence).
__device__ __forceinline__ float fast_sigmoid(float x) {
  return __builtin_amdgcn_rcpf(1.0f + __expf(-x));
}
__device__ __forceinline__ float fast_tanh(float x) {
  // tanh(x) = 1 - 2/(exp(2x)+1)
  return 1.0f - 2.0f * __builtin_amdgcn_rcpf(__expf(2.0f * x) + 1.0f);
}

__global__ __launch_bounds__(256) void arflow_gru_kernel(
    const float* __restrict__ x,   const float* __restrict__ z,
    const float* __restrict__ Wih, const float* __restrict__ Whh,
    const float* __restrict__ bih, const float* __restrict__ bhh,
    const float* __restrict__ W1,  const float* __restrict__ b1,
    const float* __restrict__ W2,  const float* __restrict__ b2,
    float* __restrict__ y_out, float* __restrict__ ld_out)
{
  __shared__ __align__(16) bf16_t sWhh[GG * WH_STRIDE];       // W_hh row-major bf16
  __shared__ __align__(16) bf16_t sW1T[32 * W1_STRIDE];       // W1^T row-major bf16
  __shared__ __align__(16) float  sHst[WAVES][RPW * HS_STRIDE];
  __shared__ __align__(16) float  sHid[WAVES][RPW * HID_STRIDE];
  __shared__ __align__(16) float  sY[WAVES][RPW * 2];
  __shared__ float sWih[GG * 2];
  __shared__ float sBih[GG], sBhh[GG], sB1[32], sB2[4];
  __shared__ float sW2T[4 * 32];

  const int tid = threadIdx.x;

  // --- cooperative weight staging / bf16 conversion -------------------------
  for (int i = tid; i < GG * HH; i += 256) {
    const int r = i >> 6, k = i & 63;
    sWhh[r * WH_STRIDE + k] = (bf16_t)Whh[i];
  }
  for (int i = tid; i < HH * 32; i += 256) {
    const int k = i >> 5, n = i & 31;                 // W1 is (64,32) row-major
    sW1T[n * W1_STRIDE + k] = (bf16_t)W1[i];
  }
  for (int i = tid; i < GG * 2; i += 256) sWih[i] = Wih[i];
  for (int i = tid; i < GG; i += 256) { sBih[i] = bih[i]; sBhh[i] = bhh[i]; }
  if (tid < 32) sB1[tid] = b1[tid];
  if (tid < 4)  sB2[tid] = b2[tid];
  for (int i = tid; i < 4 * 32; i += 256) {
    const int j = i >> 5, k = i & 31;                 // W2 is (32,4) row-major
    sW2T[j * 32 + k] = W2[k * 4 + j];
  }
  for (int i = tid; i < WAVES * RPW * 2; i += 256) (&sY[0][0])[i] = 0.0f;
  __syncthreads();

  const int lane  = tid & 31;
  const int wid   = tid >> 5;
  const int row16 = lane & 15;
  const int hf    = lane >> 4;            // lane half: 0 or 1
  const int rowBase = (blockIdx.x * WAVES + wid) * RPW;
  const int myBatch = rowBase + row16;
  const int d = hf;                       // output dim handled by this lane
  const int kb = 8 * hf;                  // per-half K base inside a B chunk

  float* hs   = sHst[wid];
  float* hidS = sHid[wid];
  float* yS   = sY[wid];

  // --- init hidden state h = z, in WMMA C-fragment layout -------------------
  // C layout: acc[v] (v=0..7) is element (M = v + 8*hf, N = nt*16 + row16).
  float hc[4][8];
  #pragma unroll
  for (int nt = 0; nt < 4; ++nt)
    #pragma unroll
    for (int v = 0; v < 8; ++v)
      hc[nt][v] = z[(size_t)(rowBase + v + 8 * hf) * HH + nt * 16 + row16];

  float yprev = 0.0f, logacc = 0.0f;
  const float* hrow = hs + row16 * HS_STRIDE;

  #pragma unroll 1
  for (int t = 0; t < TT; ++t) {
    // ---- stage h (f32) and gather bf16 A-fragments -------------------------
    #pragma unroll
    for (int nt = 0; nt < 4; ++nt)
      #pragma unroll
      for (int v = 0; v < 8; ++v)
        hs[(v + 8 * hf) * HS_STRIDE + nt * 16 + row16] = hc[nt][v];
    asm volatile("s_wait_dscnt 0" ::: "memory");

    v16bf a0 = mkA(hrow, 0, hf);
    v16bf a1 = mkA(hrow, 1, hf);

    // y_{t-1} rows needed for the K=2 input GEMM (done in VALU)
    float2 yv[8];
    #pragma unroll
    for (int v = 0; v < 8; ++v)
      yv[v] = *(const float2*)&yS[(v + 8 * hf) * 2];

    // ---- GRU gates: gh = h @ W_hh^T via WMMA, gi via VALU ------------------
    #pragma unroll
    for (int nt = 0; nt < 4; ++nt) {
      const int c = nt * 16 + row16;      // column (and W_hh row) index
      v8f accR, accZ, accN;
      {
        const float brh = sBhh[c], bzh = sBhh[64 + c], bnh = sBhh[128 + c];
        #pragma unroll
        for (int e = 0; e < 8; ++e) { accR[e] = brh; accZ[e] = bzh; accN[e] = bnh; }
      }
      accR = wmma_bf16(a0, ld_bfrag(sWhh + (size_t)c * WH_STRIDE + kb), accR);
      accR = wmma_bf16(a1, ld_bfrag(sWhh + (size_t)c * WH_STRIDE + 32 + kb), accR);
      accZ = wmma_bf16(a0, ld_bfrag(sWhh + (size_t)(64 + c) * WH_STRIDE + kb), accZ);
      accZ = wmma_bf16(a1, ld_bfrag(sWhh + (size_t)(64 + c) * WH_STRIDE + 32 + kb), accZ);
      accN = wmma_bf16(a0, ld_bfrag(sWhh + (size_t)(128 + c) * WH_STRIDE + kb), accN);
      accN = wmma_bf16(a1, ld_bfrag(sWhh + (size_t)(128 + c) * WH_STRIDE + 32 + kb), accN);

      const float wr0 = sWih[2 * c],          wr1 = sWih[2 * c + 1],          br = sBih[c];
      const float wz0 = sWih[2 * (64 + c)],   wz1 = sWih[2 * (64 + c) + 1],   bz = sBih[64 + c];
      const float wn0 = sWih[2 * (128 + c)],  wn1 = sWih[2 * (128 + c) + 1],  bn = sBih[128 + c];
      #pragma unroll
      for (int v = 0; v < 8; ++v) {
        const float gr  = accR[v] + wr0 * yv[v].x + wr1 * yv[v].y + br;
        const float gz  = accZ[v] + wz0 * yv[v].x + wz1 * yv[v].y + bz;
        const float gin = wn0 * yv[v].x + wn1 * yv[v].y + bn;
        const float rg = fast_sigmoid(gr);
        const float ug = fast_sigmoid(gz);
        const float ng = fast_tanh(gin + rg * accN[v]);
        hc[nt][v] = (1.0f - ug) * ng + ug * hc[nt][v];
      }
    }

    // ---- hid = relu(h @ W1 + b1) via WMMA ----------------------------------
    #pragma unroll
    for (int nt = 0; nt < 4; ++nt)
      #pragma unroll
      for (int v = 0; v < 8; ++v)
        hs[(v + 8 * hf) * HS_STRIDE + nt * 16 + row16] = hc[nt][v];
    asm volatile("s_wait_dscnt 0" ::: "memory");
    a0 = mkA(hrow, 0, hf);
    a1 = mkA(hrow, 1, hf);

    #pragma unroll
    for (int m = 0; m < 2; ++m) {
      const int nW = m * 16 + row16;
      v8f acc;
      const float bb = sB1[nW];
      #pragma unroll
      for (int e = 0; e < 8; ++e) acc[e] = bb;
      acc = wmma_bf16(a0, ld_bfrag(sW1T + (size_t)nW * W1_STRIDE + kb), acc);
      acc = wmma_bf16(a1, ld_bfrag(sW1T + (size_t)nW * W1_STRIDE + 32 + kb), acc);
      #pragma unroll
      for (int v = 0; v < 8; ++v)
        hidS[(v + 8 * hf) * HID_STRIDE + m * 16 + row16] = fmaxf(acc[v], 0.0f);
    }
    asm volatile("s_wait_dscnt 0" ::: "memory");

    // ---- output head (N=4): per-lane dots, scale, y update -----------------
    float lsd = sB2[d], lss = sB2[2 + d];
    const float* hr = hidS + row16 * HID_STRIDE;
    #pragma unroll
    for (int k = 0; k < 32; ++k) {
      const float hk = hr[k];
      lsd += hk * sW2T[d * 32 + k];
      lss += hk * sW2T[(2 + d) * 32 + k];
    }
    // softplus(x) = log(1+exp(x)); guard large x (exp overflow) with identity
    const float sp    = (lss > 20.0f) ? lss : __logf(1.0f + __expf(lss));
    const float scale = sp + 0.001f;
    const float xt    = x[((size_t)myBatch * TT + t) * DD + d];
    const float ynew  = yprev + lsd + scale * xt;
    y_out[((size_t)myBatch * TT + t) * DD + d] = ynew;
    logacc += __logf(scale);
    yprev = ynew;
    yS[row16 * 2 + d] = ynew;             // feeds next step's gi (K=2 GEMM)
  }

  // logabsdet[b] = sum over d of per-lane log-scale accumulators
  const float other = __shfl_down(logacc, 16, 32);
  if (hf == 0) ld_out[myBatch] = logacc + other;
}

extern "C" void kernel_launch(void* const* d_in, const int* in_sizes, int n_in,
                              void* d_out, int out_size, void* d_ws, size_t ws_size,
                              hipStream_t stream) {
  (void)in_sizes; (void)n_in; (void)d_ws; (void)ws_size; (void)out_size;
  const float* x   = (const float*)d_in[0];
  const float* z   = (const float*)d_in[1];
  const float* Wih = (const float*)d_in[2];
  const float* Whh = (const float*)d_in[3];
  const float* bih = (const float*)d_in[4];
  const float* bhh = (const float*)d_in[5];
  const float* W1  = (const float*)d_in[6];
  const float* b1  = (const float*)d_in[7];
  const float* W2  = (const float*)d_in[8];
  const float* b2  = (const float*)d_in[9];
  float* y_out  = (float*)d_out;
  float* ld_out = y_out + (size_t)BATCH * TT * DD;

  dim3 grid(BATCH / (WAVES * RPW));   // 1024 blocks of 8 waves (16 rows each)
  arflow_gru_kernel<<<grid, 256, 0, stream>>>(x, z, Wih, Whh, bih, bhh,
                                              W1, b1, W2, b2, y_out, ld_out);
}
